// CooccurrenceGraph_70738111365773
// MI455X (gfx1250) — compile-verified, matrix operand order checked by reading, not run on hardware
//
#include <hip/hip_runtime.h>
#include <hip/hip_bf16.h>

typedef __bf16 bf16_t;
typedef __attribute__((ext_vector_type(16))) __bf16 v16bf;
typedef __attribute__((ext_vector_type(8)))  __bf16 v8bf;
typedef __attribute__((ext_vector_type(4)))  __bf16 v4bf;
typedef __attribute__((ext_vector_type(8)))  float  v8f;
typedef __attribute__((ext_vector_type(4)))  float  v4f;

#define NB   2048
#define NN   80
#define DD   256
#define SX   264   // bf16 row stride for 80x256 tiles (multiple of 8)
#define SV   96    // m-stride for V^T [256][96] (padded K for attn@V)
#define SP   96    // col stride for P [80][96]
#define SS   84    // float stride for scores [80][84]

// Fragment loader: 16 bf16 elements as two contiguous 16B chunks per lane.
// 16-bit A 16x32 (and our Bt layout): lanes 0-15 hold K {0..7,16..23},
// lanes 16-31 hold K {8..15,24..31}. hiOff = (lane>=16)?8:0.
__device__ __forceinline__ v16bf load_frag(const bf16_t* row, int kbase, int hiOff) {
    v8bf c0 = *(const v8bf*)(row + kbase + hiOff);
    v8bf c1 = *(const v8bf*)(row + kbase + 16 + hiOff);
    v16bf r;
#pragma unroll
    for (int j = 0; j < 8; ++j) { r[j] = c0[j]; r[j + 8] = c1[j]; }
    return r;
}

__device__ __forceinline__ v8f wmma_bf16(v16bf a, v16bf b, v8f c) {
    return __builtin_amdgcn_wmma_f32_16x16x32_bf16(
        /*neg_a=*/false, a, /*neg_b=*/false, b,
        /*c_mod=*/(short)0, c, /*reuse_a=*/false, /*reuse_b=*/false);
}

// Convert the four 256x256 fp32 weight matrices to bf16 (row-major) in d_ws.
// These stay L2-resident (512 KB) across all 2048 attention blocks.
__global__ __launch_bounds__(256) void wconv_kernel(
    const float* __restrict__ wq, const float* __restrict__ wk,
    const float* __restrict__ wv, const float* __restrict__ wo,
    bf16_t* __restrict__ out) {
    int i = blockIdx.x * 256 + threadIdx.x;   // 0 .. 65535
    out[i]           = (bf16_t)wq[i];
    out[i +  65536]  = (bf16_t)wk[i];
    out[i + 131072]  = (bf16_t)wv[i];
    out[i + 196608]  = (bf16_t)wo[i];
}

__global__ __launch_bounds__(256) void CooccurrenceGraph_attn_kernel(
    const float* __restrict__ x, const int* __restrict__ labels,
    const float* __restrict__ bq, const float* __restrict__ bk,
    const float* __restrict__ bv, const float* __restrict__ bo,
    const float* __restrict__ cooc,
    const bf16_t* __restrict__ wqb, const bf16_t* __restrict__ wkb,
    const bf16_t* __restrict__ wvb, const bf16_t* __restrict__ wob,
    float* __restrict__ out) {
    // LDS layout (bytes):
    //  [0,42240)       xs 80x264 bf16    (aliased later by ss 80x84 f32)
    //  [42240,84480)   qs 80x264 bf16    (aliased later by os 80x264 bf16)
    //  [84480,126720)  ks 80x264 bf16  \  fp32 x staging [84480,166400)
    //  [126720,175872) vs 256x96 bf16  /  lives here before stage 1
    //  [175872,191232) ps 80x96 bf16
    __shared__ __align__(16) char smem[3 * NN * SX * 2 + DD * SV * 2 + NN * SP * 2];
    bf16_t* xs = (bf16_t*)smem;
    bf16_t* qs = (bf16_t*)(smem + NN * SX * 2);
    bf16_t* ks = (bf16_t*)(smem + 2 * NN * SX * 2);
    bf16_t* vs = (bf16_t*)(smem + 3 * NN * SX * 2);
    bf16_t* ps = (bf16_t*)(smem + 3 * NN * SX * 2 + DD * SV * 2);
    float*  ss = (float*)smem;                       // reuses xs region (stage 2+)
    bf16_t* os = qs;                                 // reuses qs region (stage 4+)
    float*  xstage = (float*)(smem + 2 * NN * SX * 2);  // 80KB fp32 staging (pre stage 1)

    const int b    = blockIdx.x;
    const int tid  = threadIdx.x;
    const int lane = tid & 31;
    const int wave = __builtin_amdgcn_readfirstlane(tid >> 5);   // scalar wave id
    const int l15  = lane & 15;
    const int hiOff = (lane & 16) ? 8 : 0;

    // ---- Stage 0a: async DMA x[b] (fp32, 80KB) into LDS staging ----
    const float* xb = x + (size_t)b * NN * DD;
    {
        unsigned sbase = (unsigned)(uintptr_t)xstage;   // LDS byte offset (addr[31:0])
        for (int i = tid; i < (NN * DD) / 4; i += 256) {      // 5120 x 16B chunks
            unsigned           ldsoff = sbase + (unsigned)i * 16u;
            unsigned long long gaddr  = (unsigned long long)(uintptr_t)(xb + i * 4);
            asm volatile("global_load_async_to_lds_b128 %0, %1, off"
                         :: "v"(ldsoff), "v"(gaddr) : "memory");
        }
        asm volatile("s_wait_asynccnt 0x0" ::: "memory");
    }
    __syncthreads();

    // ---- Stage 0b: LDS fp32 -> bf16 xs (vectorized, packed b64 stores) ----
    for (int i = tid * 4; i < NN * DD; i += 1024) {
        v4f f = *(const v4f*)(xstage + i);
        int n = i >> 8, d = i & 255;
        v4bf h;
#pragma unroll
        for (int j = 0; j < 4; ++j) h[j] = (bf16_t)f[j];
        *(v4bf*)(xs + n * SX + d) = h;
    }
    __syncthreads();   // staging fully consumed before ks/vs get written

    // ---- Stage 1: Q,K,V = x @ W^T + b  (A-frags register-blocked per row tile) ----
    for (int c = wave; c < 15; c += 8) {            // (mat, mt) combos
        int mat = c / 5, mt = c % 5;
        const bf16_t* wmat = (mat == 0) ? wqb : (mat == 1) ? wkb : wvb;
        const float*  bias = (mat == 0) ? bq  : (mat == 1) ? bk  : bv;
        const bf16_t* arow = xs + (mt * 16 + l15) * SX;
        v16bf a[8];
#pragma unroll
        for (int kt = 0; kt < 8; ++kt) a[kt] = load_frag(arow, kt * 32, hiOff);
        for (int nt = 0; nt < 16; ++nt) {
            const bf16_t* brow = wmat + (nt * 16 + l15) * DD;   // W row-major == Bt
            v8f acc = {};
#pragma unroll
            for (int kt = 0; kt < 8; ++kt)
                acc = wmma_bf16(a[kt], load_frag(brow, kt * 32, hiOff), acc);
            int dcol = nt * 16 + l15;
            float bsv = bias[dcol];
            if (mat == 2) {          // V stored transposed [d][m]
#pragma unroll
                for (int i = 0; i < 8; ++i)
                    vs[dcol * SV + mt * 16 + i + hiOff] = (bf16_t)(acc[i] + bsv);
            } else {
                bf16_t* dst = (mat == 0) ? qs : ks;
#pragma unroll
                for (int i = 0; i < 8; ++i)
                    dst[(mt * 16 + i + hiOff) * SX + dcol] = (bf16_t)(acc[i] + bsv);
            }
        }
    }
    __syncthreads();

    // ---- Stage 2: S = Q @ K^T  (25 tiles of 16x16, K = 256) ----
    for (int st = wave; st < 25; st += 8) {
        int mt = st / 5, nt = st % 5;
        const bf16_t* arow = qs + (mt * 16 + l15) * SX;
        const bf16_t* brow = ks + (nt * 16 + l15) * SX;   // K rows == Bt layout
        v8f acc = {};
#pragma unroll
        for (int kt = 0; kt < 8; ++kt) {
            v16bf a  = load_frag(arow, kt * 32, hiOff);
            v16bf bb = load_frag(brow, kt * 32, hiOff);
            acc = wmma_bf16(a, bb, acc);
        }
        int mcol = nt * 16 + l15;
#pragma unroll
        for (int i = 0; i < 8; ++i)
            ss[(mt * 16 + i + hiOff) * SS + mcol] = acc[i];
    }
    __syncthreads();

    // ---- Stage 3: modulate (cooc, label mask) + softmax; idle threads zero V pad ----
    if (tid < NN) {
        int n = tid;
        const float* coocr = cooc + n * NN;
        const int*   labr  = labels + (size_t)b * NN;
        float mx = -1e30f;
        for (int m = 0; m < NN; ++m) {
            float lm = 0.8f * (float)labr[m] + 0.2f;
            float s  = ss[n * SS + m] * 0.0625f * coocr[m] * lm;  // 1/sqrt(256)
            ss[n * SS + m] = s;
            mx = fmaxf(mx, s);
        }
        float sum = 0.0f;
        for (int m = 0; m < NN; ++m) {
            float e = __expf(ss[n * SS + m] - mx);
            ss[n * SS + m] = e;
            sum += e;
        }
        float inv = 1.0f / sum;
        for (int m = 0; m < NN; ++m) ps[n * SP + m] = (bf16_t)(ss[n * SS + m] * inv);
        for (int m = NN; m < SP; ++m) ps[n * SP + m] = (bf16_t)0.0f;  // K padding
    } else {
        // zero V^T K-padding (m = 80..95) with the otherwise-idle threads
        for (int i = tid - NN; i < DD * 16; i += 256 - NN) {
            int d = i >> 4, m = NN + (i & 15);
            vs[d * SV + m] = (bf16_t)0.0f;
        }
    }
    __syncthreads();

    // ---- Stage 4: O = P @ V  (A-frags register-blocked, K padded 80->96) ----
    for (int c = wave; c < 10; c += 8) {            // (mt, nt-half) combos
        int mt = c / 2, half = c % 2;
        const bf16_t* arow = ps + (mt * 16 + l15) * SP;
        v16bf a[3];
#pragma unroll
        for (int kt = 0; kt < 3; ++kt) a[kt] = load_frag(arow, kt * 32, hiOff);
        for (int nt = half * 8; nt < half * 8 + 8; ++nt) {
            const bf16_t* brow = vs + (nt * 16 + l15) * SV;  // V^T rows == Bt layout
            v8f acc = {};
#pragma unroll
            for (int kt = 0; kt < 3; ++kt)
                acc = wmma_bf16(a[kt], load_frag(brow, kt * 32, hiOff), acc);
            int dcol = nt * 16 + l15;
#pragma unroll
            for (int i = 0; i < 8; ++i)
                os[(mt * 16 + i + hiOff) * SX + dcol] = (bf16_t)acc[i];
        }
    }
    __syncthreads();

    // ---- Stage 5: Y = O @ Wo^T + bo -> global fp32 (A-frags register-blocked) ----
    float* outb = out + (size_t)b * NN * DD;
    for (int c = wave; c < 10; c += 8) {            // (mt, nt-half) combos
        int mt = c / 2, half = c % 2;
        const bf16_t* arow = os + (mt * 16 + l15) * SX;
        v16bf a[8];
#pragma unroll
        for (int kt = 0; kt < 8; ++kt) a[kt] = load_frag(arow, kt * 32, hiOff);
        for (int nt = half * 8; nt < half * 8 + 8; ++nt) {
            const bf16_t* brow = wob + (nt * 16 + l15) * DD;
            v8f acc = {};
#pragma unroll
            for (int kt = 0; kt < 8; ++kt)
                acc = wmma_bf16(a[kt], load_frag(brow, kt * 32, hiOff), acc);
            int dcol = nt * 16 + l15;
            float bsv = bo[dcol];
#pragma unroll
            for (int i = 0; i < 8; ++i)
                outb[(size_t)(mt * 16 + i + hiOff) * DD + dcol] = acc[i] + bsv;
        }
    }
}

extern "C" void kernel_launch(void* const* d_in, const int* in_sizes, int n_in,
                              void* d_out, int out_size, void* d_ws, size_t ws_size,
                              hipStream_t stream) {
    const float* x      = (const float*)d_in[0];
    const int*   labels = (const int*)  d_in[1];
    const float* Wq = (const float*)d_in[2];  const float* bq = (const float*)d_in[3];
    const float* Wk = (const float*)d_in[4];  const float* bk = (const float*)d_in[5];
    const float* Wv = (const float*)d_in[6];  const float* bv = (const float*)d_in[7];
    const float* Wo = (const float*)d_in[8];  const float* bo = (const float*)d_in[9];
    const float* cooc = (const float*)d_in[10];
    float* out = (float*)d_out;

    bf16_t* wbf = (bf16_t*)d_ws;   // 4 x 65536 bf16 = 512 KB, stays in L2
    wconv_kernel<<<256, 256, 0, stream>>>(Wq, Wk, Wv, Wo, wbf);
    CooccurrenceGraph_attn_kernel<<<NB, 256, 0, stream>>>(
        x, labels, bq, bk, bv, bo, cooc,
        wbf, wbf + 65536, wbf + 131072, wbf + 196608, out);
}
